// OneTransBlock_83803401879680
// MI455X (gfx1250) — matrix operand order repaired and checked by reading, not compiled, not run on hardware
//
#include <hip/hip_runtime.h>

// ---------------------------------------------------------------------------
// One transformer block (pre-norm, causal MHA, FFN with ReLU after both
// dense layers) for B=8, S=2048, D=128, F=256, H=4, Dh=32.
// All matmuls use v_wmma_f32_16x16x32_bf16 (bf16 in, f32 accumulate).
// All WMMA fragments are loaded as 2x16B (b128) accesses:
//   A-frag: two contiguous 8-elem K-chunks of a row  -> direct global loads
//   B-frag: K-contiguous run of a column             -> weights pre-transposed
// ---------------------------------------------------------------------------

typedef __bf16 bf16_t;
typedef __attribute__((ext_vector_type(16))) __bf16 v16bf;
typedef __attribute__((ext_vector_type(8)))  float  v8f;

#define LN_EPS   1e-3f
#define NEG_INF  (-1.0e9f)
#define DMODEL   128
#define SEQ      2048
#define HEADS    4
#define DHEAD    32
#define QK_SCALE 0.17677669529663687f  // 1/sqrt(32)

union Frag16 { v16bf v; uint4 q[2]; };

// float -> bf16, round to nearest even
static __device__ __forceinline__ bf16_t f2bf(float f) {
  unsigned int u = __float_as_uint(f);
  unsigned int r = (u + 0x7FFFu + ((u >> 16) & 1u)) >> 16;
  unsigned short s = (unsigned short)r;
  return __builtin_bit_cast(bf16_t, s);
}

// Load a 16-element bf16 fragment from two consecutive 16B chunks.
static __device__ __forceinline__ v16bf load_frag2(const bf16_t* p0, const bf16_t* p1) {
  Frag16 f;
  f.q[0] = *(const uint4*)p0;
  f.q[1] = *(const uint4*)p1;
  return f.v;
}

// ---------------------------------------------------------------------------
// fp32 [K,N] -> bf16 transposed [N,K] weight conversion
// ---------------------------------------------------------------------------
__global__ void cvt_t_kernel(const float* __restrict__ src, bf16_t* __restrict__ dst,
                             int K, int N) {
  int i = blockIdx.x * blockDim.x + threadIdx.x;
  if (i < K * N) {
    int kk = i / N, n = i - kk * N;       // coalesced read
    dst[(size_t)n * K + kk] = f2bf(src[i]);
  }
}

// ---------------------------------------------------------------------------
// LayerNorm over last dim (128). One wave per row; 4 elems per lane.
// ---------------------------------------------------------------------------
__global__ void ln_kernel(const float* __restrict__ x,
                          const float* __restrict__ g,
                          const float* __restrict__ b,
                          float* __restrict__ outF,
                          bf16_t* __restrict__ outB,
                          int rows) {
  const int wid  = threadIdx.x >> 5;
  const int lane = threadIdx.x & 31;
  const int row  = blockIdx.x * 8 + wid;
  if (row >= rows) return;

  const float4 xv = ((const float4*)(x + (size_t)row * DMODEL))[lane];
  float s  = xv.x + xv.y + xv.z + xv.w;
  float sq = xv.x * xv.x + xv.y * xv.y + xv.z * xv.z + xv.w * xv.w;
  #pragma unroll
  for (int off = 1; off < 32; off <<= 1) {
    s  += __shfl_xor(s,  off, 32);
    sq += __shfl_xor(sq, off, 32);
  }
  const float mu  = s * (1.0f / DMODEL);
  const float var = sq * (1.0f / DMODEL) - mu * mu;
  const float rs  = rsqrtf(var + LN_EPS);

  const float4 gv = ((const float4*)g)[lane];
  const float4 bv = ((const float4*)b)[lane];
  float o0 = (xv.x - mu) * rs * gv.x + bv.x;
  float o1 = (xv.y - mu) * rs * gv.y + bv.y;
  float o2 = (xv.z - mu) * rs * gv.z + bv.z;
  float o3 = (xv.w - mu) * rs * gv.w + bv.w;

  if (outF) {
    float4 ov = {o0, o1, o2, o3};
    ((float4*)(outF + (size_t)row * DMODEL))[lane] = ov;
  }
  // pack 4 bf16 into one 64-bit store
  unsigned short h0 = __builtin_bit_cast(unsigned short, f2bf(o0));
  unsigned short h1 = __builtin_bit_cast(unsigned short, f2bf(o1));
  unsigned short h2 = __builtin_bit_cast(unsigned short, f2bf(o2));
  unsigned short h3 = __builtin_bit_cast(unsigned short, f2bf(o3));
  uint2 pk;
  pk.x = (unsigned)h0 | ((unsigned)h1 << 16);
  pk.y = (unsigned)h2 | ((unsigned)h3 << 16);
  ((uint2*)(outB + (size_t)row * DMODEL))[lane] = pk;
}

// ---------------------------------------------------------------------------
// WMMA GEMM (no LDS): C[M,N] = A[M,K](bf16) * Wt[N,K](bf16,transposed) + bias,
// optional ReLU then optional residual add. 256 thr = 8 waves; each wave owns
// a 16(M)x64(N) tile -> 4 accumulators, 4 WMMAs per 32-K step; block tile
// 128x64. A fragment: 2 global b128 loads; each B fragment: 2 global b128.
// vt_out: write bf16 output in V-transposed [b,h,d][s] layout for attention.
// ---------------------------------------------------------------------------
__global__ void gemm_bf16_kernel(const bf16_t* __restrict__ A,
                                 const bf16_t* __restrict__ Wt,
                                 const float* __restrict__ bias,
                                 float* __restrict__ outF,
                                 bf16_t* __restrict__ outB,
                                 const float* __restrict__ residual,
                                 int M, int N, int K, int relu, int vt_out) {
  const int wid  = threadIdx.x >> 5;
  const int lane = threadIdx.x & 31;
  const int half = lane >> 4;
  const int lrow = lane & 15;

  const int m0 = blockIdx.y * 128 + wid * 16;
  const int n0 = blockIdx.x * 64;

  const bf16_t* aRow = A + (size_t)(m0 + lrow) * K;
  const bf16_t* bRow0 = Wt + (size_t)(n0 + lrow) * K + 16 * half;
  const bf16_t* bRow1 = Wt + (size_t)(n0 + 16 + lrow) * K + 16 * half;
  const bf16_t* bRow2 = Wt + (size_t)(n0 + 32 + lrow) * K + 16 * half;
  const bf16_t* bRow3 = Wt + (size_t)(n0 + 48 + lrow) * K + 16 * half;

  v8f acc0 = {}, acc1 = {}, acc2 = {}, acc3 = {};

  for (int k0 = 0; k0 < K; k0 += 32) {
    __builtin_prefetch(aRow + k0 + 32, 0, 1);
    v16bf a  = load_frag2(aRow + k0 + 16 * half, aRow + k0 + 16 * half + 8);
    v16bf b0 = load_frag2(bRow0 + k0, bRow0 + k0 + 8);
    v16bf b1 = load_frag2(bRow1 + k0, bRow1 + k0 + 8);
    v16bf b2 = load_frag2(bRow2 + k0, bRow2 + k0 + 8);
    v16bf b3 = load_frag2(bRow3 + k0, bRow3 + k0 + 8);
    acc0 = __builtin_amdgcn_wmma_f32_16x16x32_bf16(false, a, false, b0, (short)0, acc0, false, false);
    acc1 = __builtin_amdgcn_wmma_f32_16x16x32_bf16(false, a, false, b1, (short)0, acc1, false, false);
    acc2 = __builtin_amdgcn_wmma_f32_16x16x32_bf16(false, a, false, b2, (short)0, acc2, false, false);
    acc3 = __builtin_amdgcn_wmma_f32_16x16x32_bf16(false, a, false, b3, (short)0, acc3, false, false);
  }

  // D layout: reg r holds (M = r + 8*half, N = lrow) per 16-wide subtile.
  v8f* accs[4] = {&acc0, &acc1, &acc2, &acc3};
  #pragma unroll
  for (int t = 0; t < 4; ++t) {
    int col = n0 + 16 * t + lrow;
    float bcol = bias[col];
    #pragma unroll
    for (int r = 0; r < 8; ++r) {
      int row = m0 + r + (half ? 8 : 0);
      float v0 = (*accs[t])[r] + bcol;
      if (relu) v0 = fmaxf(v0, 0.0f);
      if (residual) v0 += residual[(size_t)row * N + col];
      if (outF) outF[(size_t)row * N + col] = v0;
      if (outB) {
        size_t oi = vt_out
            ? ((size_t)((row >> 11) * DMODEL + col) * SEQ + (row & (SEQ - 1)))
            : ((size_t)row * N + col);
        outB[oi] = f2bf(v0);
      }
    }
  }
}

// ---------------------------------------------------------------------------
// Causal flash attention. q/k: [B*S, 128] bf16; vt: [B*H*Dh, S] bf16 (K-major).
// One wave per 16-query tile. Keys stepped 32 at a time:
//   logits 16x32 = 2 WMMAs, online softmax, ctx += P @ V = 2 WMMAs.
// P is re-laid into A-fragment order through a per-wave LDS patch
// (16 b16 stores, 2 b128 loads per step).
// ---------------------------------------------------------------------------
__global__ void attn_kernel(const bf16_t* __restrict__ q,
                            const bf16_t* __restrict__ k,
                            const bf16_t* __restrict__ vt,
                            bf16_t* __restrict__ ctx) {
  __shared__ bf16_t fragP[4][32][16];  // [wave][lane][elem] -> A-fragment order

  const int wid  = threadIdx.x >> 5;
  const int lane = threadIdx.x & 31;
  const int half = lane >> 4;
  const int lrow = lane & 15;

  const int t  = blockIdx.x * 4 + wid;   // 0 .. B*H*(S/16)-1
  const int qt = t & (SEQ / 16 - 1);
  const int bh = t >> 7;                 // b*H + h
  const int hh = bh & (HEADS - 1);
  const int bb = bh >> 2;
  const int q0 = qt * 16;
  const size_t baseRow = (size_t)bb * SEQ;
  const int col0 = hh * DHEAD;

  // Q fragment (A operand): 16 contiguous elems at d = 16*half of its row.
  const bf16_t* qp = q + (baseRow + q0 + lrow) * DMODEL + col0;
  v16bf aQ = load_frag2(qp + 16 * half, qp + 16 * half + 8);

  float m_r[8], l_r[8];
  #pragma unroll
  for (int r = 0; r < 8; ++r) { m_r[r] = -3.0e38f; l_r[r] = 0.0f; }
  v8f c0 = {}, c1 = {};

  // LDS scatter targets for this lane's softmax values (A-frag inverse map):
  // element (m, kk): lane' = 16*((kk>>3)&1) + m, elem = (kk>>4)*8 + (kk&7).
  const int pl = 16 * ((lrow >> 3) & 1);  // lane' base from kk=lrow / kk=16+lrow
  const int pe = lrow & 7;

  for (int kb0 = 0; kb0 < q0 + 16; kb0 += 32) {
    // K^T fragments (B operand): 16 K-contiguous elems of key row at d=16*half.
    const bf16_t* kp0 = k + (baseRow + kb0 + lrow) * DMODEL + col0 + 16 * half;
    const bf16_t* kp1 = k + (baseRow + kb0 + 16 + lrow) * DMODEL + col0 + 16 * half;
    v16bf bk0 = load_frag2(kp0, kp0 + 8);
    v16bf bk1 = load_frag2(kp1, kp1 + 8);

    v8f z = {};
    v8f s0 = __builtin_amdgcn_wmma_f32_16x16x32_bf16(false, aQ, false, bk0, (short)0, z, false, false);
    v8f s1 = __builtin_amdgcn_wmma_f32_16x16x32_bf16(false, aQ, false, bk1, (short)0, z, false, false);

    #pragma unroll
    for (int r = 0; r < 8; ++r) {
      int rowg = q0 + r + (half ? 8 : 0);
      float p0 = s0[r] * QK_SCALE + ((kb0 + lrow)      <= rowg ? 0.0f : NEG_INF);
      float p1 = s1[r] * QK_SCALE + ((kb0 + 16 + lrow) <= rowg ? 0.0f : NEG_INF);
      float tm = fmaxf(p0, p1);
      #pragma unroll
      for (int off = 1; off < 16; off <<= 1) tm = fmaxf(tm, __shfl_xor(tm, off, 32));
      float mn = fmaxf(m_r[r], tm);
      float e0 = __expf(p0 - mn);
      float e1 = __expf(p1 - mn);
      float rs = e0 + e1;
      #pragma unroll
      for (int off = 1; off < 16; off <<= 1) rs += __shfl_xor(rs, off, 32);
      float sc = __expf(m_r[r] - mn);
      l_r[r] = l_r[r] * sc + rs;
      m_r[r] = mn;
      c0[r] *= sc;
      c1[r] *= sc;
      int m = r + (half ? 8 : 0);
      fragP[wid][pl + m][pe]     = f2bf(e0);  // kk = lrow
      fragP[wid][pl + m][8 + pe] = f2bf(e1);  // kk = 16 + lrow
    }

    // P as A operand: contiguous per-lane fragment -> 2 x ds_load_b128.
    v16bf aP = load_frag2(&fragP[wid][lane][0], &fragP[wid][lane][8]);

    // V fragments (B operand) from K-major vt: keys 8*half..+7 and 16+8*half..+7 per d-col.
    const bf16_t* vp0 = vt + (size_t)(bh * DHEAD + lrow) * SEQ + kb0;
    const bf16_t* vp1 = vt + (size_t)(bh * DHEAD + 16 + lrow) * SEQ + kb0;
    v16bf bv0 = load_frag2(vp0 + 8 * half, vp0 + 16 + 8 * half);
    v16bf bv1 = load_frag2(vp1 + 8 * half, vp1 + 16 + 8 * half);

    c0 = __builtin_amdgcn_wmma_f32_16x16x32_bf16(false, aP, false, bv0, (short)0, c0, false, false);
    c1 = __builtin_amdgcn_wmma_f32_16x16x32_bf16(false, aP, false, bv1, (short)0, c1, false, false);
  }

  #pragma unroll
  for (int r = 0; r < 8; ++r) {
    float inv = 1.0f / l_r[r];
    int row = q0 + r + (half ? 8 : 0);
    bf16_t* cp = ctx + (baseRow + row) * DMODEL + col0;
    cp[lrow]      = f2bf(c0[r] * inv);
    cp[16 + lrow] = f2bf(c1[r] * inv);
  }
}

// ---------------------------------------------------------------------------
// Host-side launch
// ---------------------------------------------------------------------------
extern "C" void kernel_launch(void* const* d_in, const int* in_sizes, int n_in,
                              void* d_out, int out_size, void* d_ws, size_t ws_size,
                              hipStream_t stream) {
  const float* x     = (const float*)d_in[0];
  const float* ln0_g = (const float*)d_in[1];
  const float* ln0_b = (const float*)d_in[2];
  const float* ln1_g = (const float*)d_in[3];
  const float* ln1_b = (const float*)d_in[4];
  const float* wq = (const float*)d_in[5];  const float* bq = (const float*)d_in[6];
  const float* wk = (const float*)d_in[7];  const float* bk = (const float*)d_in[8];
  const float* wv = (const float*)d_in[9];  const float* bv = (const float*)d_in[10];
  const float* wo = (const float*)d_in[11]; const float* bo = (const float*)d_in[12];
  const float* w1 = (const float*)d_in[13]; const float* b1 = (const float*)d_in[14];
  const float* w2 = (const float*)d_in[15]; const float* b2 = (const float*)d_in[16];

  const int M = 8 * SEQ;   // 16384 token rows
  const int D = DMODEL;    // 128
  const int F = 256;

  char* p = (char*)d_ws;
  auto alloc = [&](size_t bytes) {
    void* r = (void*)p;
    p += (bytes + 255) & ~(size_t)255;
    return r;
  };
  bf16_t* wq_t = (bf16_t*)alloc((size_t)D * D * 2);   // [N=D][K=D]
  bf16_t* wk_t = (bf16_t*)alloc((size_t)D * D * 2);
  bf16_t* wv_t = (bf16_t*)alloc((size_t)D * D * 2);
  bf16_t* wo_t = (bf16_t*)alloc((size_t)D * D * 2);
  bf16_t* w1_t = (bf16_t*)alloc((size_t)F * D * 2);   // [N=F][K=D]
  bf16_t* w2_t = (bf16_t*)alloc((size_t)D * F * 2);   // [N=D][K=F]
  float*  h_f  = (float*) alloc((size_t)M * D * 4);   // LN0 output (residual)
  bf16_t* h_b  = (bf16_t*)alloc((size_t)M * D * 2);
  bf16_t* q_b  = (bf16_t*)alloc((size_t)M * D * 2);
  bf16_t* k_b  = (bf16_t*)alloc((size_t)M * D * 2);
  bf16_t* vt_b = (bf16_t*)alloc((size_t)M * D * 2);   // V transposed [b,h,d][s]
  bf16_t* ctx_b= (bf16_t*)alloc((size_t)M * D * 2);
  float*  x1_f = (float*) alloc((size_t)M * D * 4);   // attn residual output
  bf16_t* h2_b = (bf16_t*)alloc((size_t)M * D * 2);
  bf16_t* f_b  = (bf16_t*)alloc((size_t)M * F * 2);
  (void)ws_size; (void)n_in; (void)in_sizes; (void)out_size;

  // 1) weights -> bf16, transposed to [N][K]
  cvt_t_kernel<<<(D * D + 255) / 256, 256, 0, stream>>>(wq, wq_t, D, D);
  cvt_t_kernel<<<(D * D + 255) / 256, 256, 0, stream>>>(wk, wk_t, D, D);
  cvt_t_kernel<<<(D * D + 255) / 256, 256, 0, stream>>>(wv, wv_t, D, D);
  cvt_t_kernel<<<(D * D + 255) / 256, 256, 0, stream>>>(wo, wo_t, D, D);
  cvt_t_kernel<<<(D * F + 255) / 256, 256, 0, stream>>>(w1, w1_t, D, F);
  cvt_t_kernel<<<(F * D + 255) / 256, 256, 0, stream>>>(w2, w2_t, F, D);

  // 2) LN0: x -> h (f32 residual + bf16 GEMM operand)
  ln_kernel<<<M / 8, 256, 0, stream>>>(x, ln0_g, ln0_b, h_f, h_b, M);

  // 3) Q/K/V projections (V written K-major for attention)
  gemm_bf16_kernel<<<dim3(D / 64, M / 128), 256, 0, stream>>>(h_b, wq_t, bq, nullptr, q_b,  nullptr, M, D, D, 0, 0);
  gemm_bf16_kernel<<<dim3(D / 64, M / 128), 256, 0, stream>>>(h_b, wk_t, bk, nullptr, k_b,  nullptr, M, D, D, 0, 0);
  gemm_bf16_kernel<<<dim3(D / 64, M / 128), 256, 0, stream>>>(h_b, wv_t, bv, nullptr, vt_b, nullptr, M, D, D, 0, 1);

  // 4) causal attention -> ctx
  attn_kernel<<<(8 * HEADS * (SEQ / 16)) / 4, 128, 0, stream>>>(q_b, k_b, vt_b, ctx_b);

  // 5) output projection + residual (origin = h)
  gemm_bf16_kernel<<<dim3(D / 64, M / 128), 256, 0, stream>>>(ctx_b, wo_t, bo, x1_f, nullptr, h_f, M, D, D, 0, 0);

  // 6) LN1 -> h2 (bf16)
  ln_kernel<<<M / 8, 256, 0, stream>>>(x1_f, ln1_g, ln1_b, nullptr, h2_b, M);

  // 7) FFN: relu(h2@w1+b1) -> f ; relu(f@w2+b2) + x1 -> out
  gemm_bf16_kernel<<<dim3(F / 64, M / 128), 256, 0, stream>>>(h2_b, w1_t, b1, nullptr, f_b, nullptr, M, F, D, 1, 0);
  gemm_bf16_kernel<<<dim3(D / 64, M / 128), 256, 0, stream>>>(f_b, w2_t, b2, (float*)d_out, nullptr, x1_f, M, D, F, 1, 0);
}
